// multi_triples_lstm_67362267070530
// MI455X (gfx1250) — matrix operand order repaired and checked by reading, not compiled, not run on hardware
//
#include <hip/hip_runtime.h>

typedef __attribute__((ext_vector_type(16))) _Float16 v16h;
typedef __attribute__((ext_vector_type(8)))  float    v8f;

#define B_TOTAL 32768
#define T_STEPS 32
#define OBJ_V 179
#define PRED_V 46

union AFrag {
    v16h v;
    _Float16 h[16];
    uint4 q[2];
};

__device__ __forceinline__ float fast_tanh(float x) {
#if __has_builtin(__builtin_amdgcn_tanhf)
    return __builtin_amdgcn_tanhf(x);        // v_tanh_f32 (gfx1250 trans op)
#else
    float e = __builtin_amdgcn_exp2f(2.8853900817779268f * x);
    return 1.0f - 2.0f * __builtin_amdgcn_rcpf(e + 1.0f);
#endif
}
__device__ __forceinline__ float fast_sigmoid(float x) {
#if __has_builtin(__builtin_amdgcn_tanhf)
    // sigmoid(x) = 0.5*tanh(x/2) + 0.5  -> 1 trans + 1 fma
    return __builtin_fmaf(0.5f, __builtin_amdgcn_tanhf(0.5f * x), 0.5f);
#else
    float e = __builtin_amdgcn_exp2f(-1.4426950408889634f * x);
    return __builtin_amdgcn_rcpf(1.0f + e);
#endif
}

// Wave-local ordering point: each wave's s_h region is private, and LDS ops from a
// single wave are processed in order by the LDS pipe (CDNA5 ISA §7.3), so a
// compiler-only fence is sufficient -- no block barrier, no hardware wait.
__device__ __forceinline__ void lds_wave_fence() {
    asm volatile("" ::: "memory");
}

__global__ __launch_bounds__(256)
void lstm_wmma_kernel(const int* __restrict__ objs,
                      const float* __restrict__ boxes,
                      const int* __restrict__ preds,
                      const int* __restrict__ subj,
                      const float* __restrict__ obj_emb,
                      const float* __restrict__ pred_emb,
                      const float* __restrict__ W_ih,
                      const float* __restrict__ W_hh,
                      const float* __restrict__ b_ih,
                      const float* __restrict__ b_hh,
                      const float* __restrict__ fc2_W,
                      const float* __restrict__ fc2_b,
                      float* __restrict__ out)
{
    // B-matrix weight fragments in WMMA lane layout: [kc*8+nt][lane][16 halfs]
    __shared__ __align__(32) _Float16 s_bfrag[16 * 32 * 16];   // 16 KB
    // per-wave h buffer: 16 rows x 32 units, row-major, natural unit order
    __shared__ __align__(16) _Float16 s_h[8 * 16 * 32];        // 8 KB
    __shared__ float s_obj[OBJ_V * 5];
    __shared__ float s_pred[PRED_V * 5];
    __shared__ float s_bias[128];
    __shared__ float s_fc2w[128];
    __shared__ float s_fc2b[4];

    const int tid = threadIdx.x;

    // ---------------- block prologue: stage tables + pre-swizzled weights -------------
    for (int i = tid; i < OBJ_V * 5; i += 256)  s_obj[i]  = obj_emb[i];
    for (int i = tid; i < PRED_V * 5; i += 256) s_pred[i] = pred_emb[i];
    if (tid < 128) s_fc2w[tid] = fc2_W[tid];
    if (tid < 4)   s_fc2b[tid] = fc2_b[tid];
    if (tid < 128) {
        // gate column N = tid -> permuted hidden unit; fold both biases
        int g = tid >> 5, n = tid & 31;
        int rho = g * 32 + 2 * (n & 15) + (n >> 4);
        s_bias[tid] = b_ih[rho] + b_hh[rho];
    }
    // Build W fragments: K = [x(0..15) | h(16..47) | pad(48..63)], N = 128 gate cols.
    // B-fragment layout: lane L holds column N = L%16, K = 16*(e/8) + 8*(L/16) + e%8 for element e.
    for (int idx = tid; idx < 16 * 32 * 16; idx += 256) {
        int fi = idx >> 9;            // fragment: kc*8 + nt
        int L  = (idx >> 4) & 31;     // lane
        int e  = idx & 15;            // element within lane
        int kc = fi >> 3, nt = fi & 7;
        int kk = kc * 32 + ((e >> 3) << 4) + ((L >> 4) << 3) + (e & 7);   // overall K 0..63
        int Ng = nt * 16 + (L & 15);              // gate column 0..127
        int g = Ng >> 5, n = Ng & 31;
        int rho = g * 32 + 2 * (n & 15) + (n >> 4);  // original torch gate row
        float v = 0.0f;
        if (kk < 16)      v = W_ih[rho * 16 + kk];
        else if (kk < 48) v = W_hh[rho * 32 + (kk - 16)];
        s_bfrag[idx] = (_Float16)v;
    }
    // zero h state
    for (int i = tid; i < (8 * 16 * 32) / 2; i += 256) ((unsigned int*)s_h)[i] = 0u;
    __syncthreads();   // the only block barrier: weights/tables now visible to all waves

    const int lane   = tid & 31;
    const int half   = lane >> 4;     // A/K half of the lane
    const int l      = lane & 15;
    const int waveId = tid >> 5;
    const int rowBase = blockIdx.x * 128 + waveId * 16;
    const int bf = rowBase + l;       // batch row this lane feeds into A

    // Pre-splatted bias accumulator-init vectors (saves 64 v_mov per timestep)
    v8f biasv[8];
#pragma unroll
    for (int nt = 0; nt < 8; ++nt) {
        float bv = s_bias[nt * 16 + l];
#pragma unroll
        for (int m = 0; m < 8; ++m) biasv[nt][m] = bv;
    }

    // c-state: 8 D-rows (M = 8*half + m) x 2 adjacent hidden units (2l, 2l+1)
    float c_st[16];
#pragma unroll
    for (int i = 0; i < 16; ++i) c_st[i] = 0.0f;

    const uint4* hrow = (const uint4*)&s_h[waveId * 512 + l * 32];  // this lane's A row
    unsigned int* hstoreU = ((unsigned int*)s_h) + waveId * 256;

    for (int t = 0; t < T_STEPS; ++t) {
        AFrag a0, a1;
        // ---- per-lane feature gather (x part of A, K 0..15 split across lane halves)
        {
            float f[8];
            if (half == 0) {          // feats 0..7: obj_emb(5) + pred_emb[0..2]
                int oid = objs[bf * T_STEPS + t];
                int pid = preds[bf * T_STEPS + t];
                f[0] = s_obj[oid * 5 + 0]; f[1] = s_obj[oid * 5 + 1];
                f[2] = s_obj[oid * 5 + 2]; f[3] = s_obj[oid * 5 + 3];
                f[4] = s_obj[oid * 5 + 4];
                f[5] = s_pred[pid * 5 + 0]; f[6] = s_pred[pid * 5 + 1];
                f[7] = s_pred[pid * 5 + 2];
            } else {                  // feats 8..15: pred_emb[3..4] + onehot(2) + boxes(4)
                int pid = preds[bf * T_STEPS + t];
                int sb  = subj[bf * T_STEPS + t];
                const float* bx = &boxes[(bf * T_STEPS + t) * 4];
                f[0] = s_pred[pid * 5 + 3]; f[1] = s_pred[pid * 5 + 4];
                f[2] = (sb == 0) ? 1.0f : 0.0f;
                f[3] = (sb == 1) ? 1.0f : 0.0f;
                f[4] = bx[0]; f[5] = bx[1]; f[6] = bx[2]; f[7] = bx[3];
            }
#pragma unroll
            for (int j = 0; j < 8; ++j) a0.h[j] = (_Float16)f[j];
        }
        // ---- h part of A from LDS (units 0..15 in chunk0, 16..31 in chunk1)
        a0.q[1] = hrow[half];
        a1.q[0] = hrow[2 + half];
#pragma unroll
        for (int j = 8; j < 16; ++j) a1.h[j] = (_Float16)0.0f;   // K pad 48..63

        // ---- gates = A x W via 16 WMMAs, bias-initialized f32 accumulators
        v8f acc[8];
#pragma unroll
        for (int nt = 0; nt < 8; ++nt) {
            v16h b0 = *(const v16h*)&s_bfrag[(nt * 32 + lane) * 16];
            v8f c = __builtin_amdgcn_wmma_f32_16x16x32_f16(false, a0.v, false, b0,
                                                           (short)0, biasv[nt],
                                                           false, false);
            v16h b1 = *(const v16h*)&s_bfrag[((8 + nt) * 32 + lane) * 16];
            c = __builtin_amdgcn_wmma_f32_16x16x32_f16(false, a1.v, false, b1,
                                                       (short)0, c, false, false);
            acc[nt] = c;
        }

        // ---- elementwise LSTM cell update; this lane owns rows 8*half+m,
        // units {2l, 2l+1} (column permutation) -> packed b32 store
#pragma unroll
        for (int m = 0; m < 8; ++m) {
            unsigned int pk = 0;
#pragma unroll
            for (int s = 0; s < 2; ++s) {
                float iv = fast_sigmoid(acc[0 + s][m]);
                float fv = fast_sigmoid(acc[2 + s][m]);
                float gv = fast_tanh(acc[4 + s][m]);
                float ov = fast_sigmoid(acc[6 + s][m]);
                float cn = fv * c_st[m * 2 + s] + iv * gv;
                c_st[m * 2 + s] = cn;
                float hv = ov * fast_tanh(cn);
                _Float16 hh = (_Float16)hv;
                unsigned short us;
                __builtin_memcpy(&us, &hh, 2);
                pk |= ((unsigned int)us) << (16 * s);
            }
            hstoreU[(8 * half + m) * 16 + l] = pk;
        }
        // wave-private region + in-order LDS pipe: compiler fence only, no barrier
        lds_wave_fence();
    }

    // ---------------- epilogue: out = clamp(h @ fc2_W^T + b, 0, 1) ------------------
    const _Float16* hr = &s_h[waveId * 512 + l * 32];
#pragma unroll
    for (int oi = 0; oi < 2; ++oi) {
        int oo = half + 2 * oi;
        float a = s_fc2b[oo];
#pragma unroll
        for (int u = 0; u < 32; ++u) a += s_fc2w[oo * 32 + u] * (float)hr[u];
        a = fminf(fmaxf(a, 0.0f), 1.0f);
        out[(rowBase + l) * 4 + oo] = a;
    }
}

extern "C" void kernel_launch(void* const* d_in, const int* in_sizes, int n_in,
                              void* d_out, int out_size, void* d_ws, size_t ws_size,
                              hipStream_t stream) {
    const int*   objs     = (const int*)d_in[0];
    const float* boxes    = (const float*)d_in[1];
    const int*   preds    = (const int*)d_in[2];
    const int*   subj     = (const int*)d_in[3];
    // d_in[4] = target_object (unused by reference forward)
    const float* obj_emb  = (const float*)d_in[5];
    const float* pred_emb = (const float*)d_in[6];
    const float* W_ih     = (const float*)d_in[7];
    const float* W_hh     = (const float*)d_in[8];
    const float* b_ih     = (const float*)d_in[9];
    const float* b_hh     = (const float*)d_in[10];
    const float* fc2_W    = (const float*)d_in[11];
    const float* fc2_b    = (const float*)d_in[12];
    float* out = (float*)d_out;

    dim3 grid(B_TOTAL / 128);   // 8 waves x 16 rows per block
    dim3 block(256);
    hipLaunchKernelGGL(lstm_wmma_kernel, grid, block, 0, stream,
                       objs, boxes, preds, subj, obj_emb, pred_emb,
                       W_ih, W_hh, b_ih, b_hh, fc2_W, fc2_b, out);
}
